// To1DFreqEmbedding_87978110091355
// MI455X (gfx1250) — compile-verified, matrix operand order checked by reading, not run on hardware
//
#include <hip/hip_runtime.h>

typedef __attribute__((ext_vector_type(16))) _Float16 v16h;
typedef __attribute__((ext_vector_type(8)))  _Float16 v8h;
typedef __attribute__((ext_vector_type(8)))  float    v8f;
typedef __attribute__((ext_vector_type(4)))  float    v4f;

#define TPB   256
#define ROWS  128      // samples (t values) per block
#define D_DIM 128
#define K_DIM 1024
#define T_DIM 512
#define ASTR  72       // halves per staged-A row (64 data + 8 pad)
#define HSTR  132      // floats per row of f32 LDS tiles (128 + 4 pad, keeps 16B align)

__device__ __forceinline__ float sigm(float x)      { return 1.0f / (1.0f + __expf(-x)); }
__device__ __forceinline__ float tanh_fast(float x) { return 1.0f - 2.0f / (1.0f + __expf(2.0f * x)); }

// K x 16 B-operand fragment (or 16 x K A-operand from an f32 row-major tile):
// lane holds row/col (ln) of W; K pattern = {k0..k0+7, k0+16..k0+23} per ISA 16-bit layout.
__device__ __forceinline__ v16h load_f32_frag(const float* __restrict__ p) {
  v4f a0 = *(const v4f*)(p);
  v4f a1 = *(const v4f*)(p + 4);
  v4f a2 = *(const v4f*)(p + 16);
  v4f a3 = *(const v4f*)(p + 20);
  v16h r;
#pragma unroll
  for (int j = 0; j < 4; ++j) {
    r[j]      = (_Float16)a0[j];
    r[j + 4]  = (_Float16)a1[j];
    r[j + 8]  = (_Float16)a2[j];
    r[j + 12] = (_Float16)a3[j];
  }
  return r;
}

// A fragment from the f16 staged tile (two ds_load_b128 per lane)
__device__ __forceinline__ v16h load_f16_frag(const _Float16* p) {
  v8h lo  = *(const v8h*)(p);
  v8h hi8 = *(const v8h*)(p + 16);
  v16h r;
#pragma unroll
  for (int j = 0; j < 8; ++j) { r[j] = lo[j]; r[j + 8] = hi8[j]; }
  return r;
}

#define WMMA_F16(a, b, c) \
  __builtin_amdgcn_wmma_f32_16x16x32_f16(false, (a), false, (b), (short)0, (c), false, false)

__global__ __launch_bounds__(TPB)
void fused_freq_emb_kernel(
    const float* __restrict__ x,      const float* __restrict__ fc1_w, const float* __restrict__ fc1_b,
    const float* __restrict__ wih_f,  const float* __restrict__ whh_f,
    const float* __restrict__ bih_f,  const float* __restrict__ bhh_f,
    const float* __restrict__ wih_b,  const float* __restrict__ whh_b,
    const float* __restrict__ bih_b,  const float* __restrict__ bhh_b,
    const float* __restrict__ lout_w, const float* __restrict__ lout_b,
    const float* __restrict__ emb_w1, const float* __restrict__ emb_b1,
    const float* __restrict__ emb_w2, const float* __restrict__ emb_b2,
    float* __restrict__ partial)
{
  extern __shared__ char smem[];
  _Float16* Ast = (_Float16*)smem;                          // 128*72*2   = 18432 B
  float*    Hs  = (float*)(smem + ROWS * ASTR * 2);         // 128*132*4  = 67584 B
  float*    Ys  = Hs + ROWS * HSTR;                         // 67584 B
  float*    colsum = Ys + ROWS * HSTR;                      // 512 B

  const int tid  = threadIdx.x;
  const int lane = tid & 31;
  const int wave = tid >> 5;        // 0..7  -> d-strip [16w, 16w+16)
  const int ln   = lane & 15;
  const int hi   = lane >> 4;
  const int kq   = hi * 8;          // K sub-offset per half-wave
  const int g    = blockIdx.x;      // 0..255
  const int b    = g >> 2;
  const int t0   = (g & 3) * ROWS;
  const int d1   = wave * 16 + ln;  // this lane's output column d

  // zero the y-accumulation tile and column sums (fenced by the first barrier below)
  for (int i = tid; i < ROWS * HSTR; i += TPB) Ys[i] = 0.0f;
  if (tid < D_DIM) colsum[tid] = 0.0f;

  // ---------------- GEMM1: h[m, d] = sum_f x[b, f, t0+m] * fc1_w[d, f] ----------------
  v8f acc[8] = {};
  const float* xb = x + (size_t)b * (K_DIM * T_DIM) + t0;

#pragma unroll 1
  for (int kc = 0; kc < K_DIM / 64; ++kc) {
    const int f0 = kc * 64;
    __syncthreads();
    // cooperative transpose-stage: 64 f-rows x 128 t, f32 -> f16
#pragma unroll
    for (int i = 0; i < 4; ++i) {
      const int u  = i * TPB + tid;   // 0..1023
      const int kp = u >> 5;          // f-pair index 0..31
      const int tq = u & 31;          // t-quad index 0..31
      const float* p0 = xb + (size_t)(f0 + 2 * kp) * T_DIM + 4 * tq;
      v4f g0 = *(const v4f*)p0;
      v4f g1 = *(const v4f*)(p0 + T_DIM);
#pragma unroll
      for (int j = 0; j < 4; ++j) {
        union { _Float16 h[2]; unsigned u32; } pk;
        pk.h[0] = (_Float16)g0[j];
        pk.h[1] = (_Float16)g1[j];
        *(unsigned*)&Ast[(4 * tq + j) * ASTR + 2 * kp] = pk.u32;
      }
    }
    if (kc + 1 < K_DIM / 64)  // hint next x tile toward the caches
      __builtin_prefetch(xb + (size_t)(f0 + 64) * T_DIM + tid * 32, 0, 1);
    __syncthreads();

#pragma unroll
    for (int ks = 0; ks < 64; ks += 32) {
      v16h bf = load_f32_frag(fc1_w + (size_t)d1 * K_DIM + (f0 + ks + kq));
#pragma unroll
      for (int mt = 0; mt < 8; ++mt) {
        v16h af = load_f16_frag(Ast + (mt * 16 + ln) * ASTR + ks + kq);
        acc[mt] = WMMA_F16(af, bf, acc[mt]);
      }
    }
  }

  {
    const float bias1 = fc1_b[d1];
#pragma unroll
    for (int mt = 0; mt < 8; ++mt)
#pragma unroll
      for (int j = 0; j < 8; ++j)
        Hs[(mt * 16 + j + 8 * hi) * HSTR + d1] = acc[mt][j] + bias1;
  }
  __syncthreads();

  // ---------------- bidirectional LSTM over the 128 channels (H = 1) ----------------
  {
    const int  m    = tid & 127;
    const bool back = tid >= 128;
    const float* wih = back ? wih_b : wih_f;
    const float* whh = back ? whh_b : whh_f;
    const float* bi  = back ? bih_b : bih_f;
    const float* bh  = back ? bhh_b : bhh_f;
    const float wi0 = wih[0], wi1 = wih[1], wi2 = wih[2], wi3 = wih[3];
    const float wh0 = whh[0], wh1 = whh[1], wh2 = whh[2], wh3 = whh[3];
    const float b0 = bi[0] + bh[0], b1 = bi[1] + bh[1];
    const float b2 = bi[2] + bh[2], b3 = bi[3] + bh[3];
    const float wy = back ? lout_w[1] : lout_w[0];
    const float by = back ? lout_b[0] : 0.0f;

    float h = 0.0f, c = 0.0f;
    for (int s = 0; s < D_DIM; ++s) {
      const int t = back ? (D_DIM - 1 - s) : s;
      const float xt = Hs[m * HSTR + t];
      const float zi = fmaf(wh0, h, fmaf(wi0, xt, b0));
      const float zf = fmaf(wh1, h, fmaf(wi1, xt, b1));
      const float zg = fmaf(wh2, h, fmaf(wi2, xt, b2));
      const float zo = fmaf(wh3, h, fmaf(wi3, xt, b3));
      c = fmaf(sigm(zf), c, sigm(zi) * tanh_fast(zg));
      h = sigm(zo) * tanh_fast(c);
      atomicAdd(&Ys[m * HSTR + t], fmaf(wy, h, by));  // ds_add_f32: y = w0*fwd + w1*bwd + b
    }
  }
  __syncthreads();

  // ---------------- emb1: a1 = leaky(y @ emb_w1.T + emb_b1) ----------------
  v8f a1[8] = {};
#pragma unroll
  for (int kc = 0; kc < 4; ++kc) {
    const int k0 = kc * 32;
    v16h bf = load_f32_frag(emb_w1 + (size_t)d1 * D_DIM + k0 + kq);
#pragma unroll
    for (int mt = 0; mt < 8; ++mt) {
      v16h af = load_f32_frag(Ys + (mt * 16 + ln) * HSTR + k0 + kq);
      a1[mt] = WMMA_F16(af, bf, a1[mt]);
    }
  }
  {
    const float be1 = emb_b1[d1];
#pragma unroll
    for (int mt = 0; mt < 8; ++mt)
#pragma unroll
      for (int j = 0; j < 8; ++j) {
        float v = a1[mt][j] + be1;
        v = v >= 0.0f ? v : 0.2f * v;
        Hs[(mt * 16 + j + 8 * hi) * HSTR + d1] = v;   // Hs reused as a1 tile
      }
  }
  __syncthreads();

  // ---------------- emb2 + mean over t ----------------
  v8f a2[8] = {};
#pragma unroll
  for (int kc = 0; kc < 4; ++kc) {
    const int k0 = kc * 32;
    v16h bf = load_f32_frag(emb_w2 + (size_t)d1 * D_DIM + k0 + kq);
#pragma unroll
    for (int mt = 0; mt < 8; ++mt) {
      v16h af = load_f32_frag(Hs + (mt * 16 + ln) * HSTR + k0 + kq);
      a2[mt] = WMMA_F16(af, bf, a2[mt]);
    }
  }
  {
    const float be2 = emb_b2[d1];
    float s = 0.0f;
#pragma unroll
    for (int mt = 0; mt < 8; ++mt)
#pragma unroll
      for (int j = 0; j < 8; ++j)
        s += a2[mt][j] + be2;
    atomicAdd(&colsum[d1], s);   // lanes L and L+16 hold complementary m halves of column d
  }
  __syncthreads();

  if (tid < D_DIM)
    partial[(size_t)g * D_DIM + tid] = colsum[tid] * (1.0f / (float)T_DIM);
}

__global__ void reduce_partials_kernel(const float* __restrict__ partial, float* __restrict__ out) {
  const int i = blockIdx.x * blockDim.x + threadIdx.x;  // 0..8191
  const int b = i >> 7;
  const int d = i & 127;
  const float* p = partial + (size_t)(b * 4) * D_DIM + d;
  out[i] = p[0] + p[D_DIM] + p[2 * D_DIM] + p[3 * D_DIM];
}

extern "C" void kernel_launch(void* const* d_in, const int* in_sizes, int n_in,
                              void* d_out, int out_size, void* d_ws, size_t ws_size,
                              hipStream_t stream) {
  const float* x      = (const float*)d_in[0];
  const float* fc1_w  = (const float*)d_in[1];
  const float* fc1_b  = (const float*)d_in[2];
  const float* wih_f  = (const float*)d_in[3];
  const float* whh_f  = (const float*)d_in[4];
  const float* bih_f  = (const float*)d_in[5];
  const float* bhh_f  = (const float*)d_in[6];
  const float* wih_b  = (const float*)d_in[7];
  const float* whh_b  = (const float*)d_in[8];
  const float* bih_b  = (const float*)d_in[9];
  const float* bhh_b  = (const float*)d_in[10];
  const float* lout_w = (const float*)d_in[11];
  const float* lout_b = (const float*)d_in[12];
  const float* emb_w1 = (const float*)d_in[13];
  const float* emb_b1 = (const float*)d_in[14];
  const float* emb_w2 = (const float*)d_in[15];
  const float* emb_b2 = (const float*)d_in[16];
  float* partial = (float*)d_ws;             // 256 * 128 floats = 128 KB
  float* out     = (float*)d_out;            // (64, 128)

  const size_t shmem = (size_t)ROWS * ASTR * 2           // f16 A stage
                     + (size_t)ROWS * HSTR * 4 * 2       // Hs + Ys f32 tiles
                     + (size_t)D_DIM * 4;                // colsum
  (void)hipFuncSetAttribute(reinterpret_cast<const void*>(fused_freq_emb_kernel),
                            hipFuncAttributeMaxDynamicSharedMemorySize, (int)shmem);

  fused_freq_emb_kernel<<<256, TPB, shmem, stream>>>(
      x, fc1_w, fc1_b, wih_f, whh_f, bih_f, bhh_f,
      wih_b, whh_b, bih_b, bhh_b, lout_w, lout_b,
      emb_w1, emb_b1, emb_w2, emb_b2, partial);

  reduce_partials_kernel<<<(64 * D_DIM) / TPB, TPB, 0, stream>>>(partial, out);
}
